// HiveGNN_75780402970781
// MI455X (gfx1250) — matrix operand order repaired and checked by reading, not compiled, not run on hardware
//
#include <hip/hip_runtime.h>

typedef __attribute__((ext_vector_type(16))) __bf16 v16bf;
typedef __attribute__((ext_vector_type(8)))  __bf16 v8bf;
typedef __attribute__((ext_vector_type(8)))  float  v8f;

#define HD    256      // hidden width H
#define NNODE 64       // nodes per graph N
#define BB    64       // batch B
#define NNTOT 4096     // B*N
#define FIN   64       // input features
#define EDGES 32768    // B*E_PER

// ---------------------------------------------------------------------------
// Fragment loading per CDNA5 ISA 7.12.2:
//   A (16x32 bf16): lane L holds row L%16; elems 0..7 -> K=k0+8*(L/16)+0..7,
//                   elems 8..15 -> K=k0+16+8*(L/16)+0..7  (two contiguous v8bf)
//   B (32x16 bf16): lane L holds col L%16 of a K x 16 tile; elems 0..7 ->
//                   K=k0+8*(L/16)+0..7, elems 8..15 -> K=k0+16+8*(L/16)+0..7 (as A; row of the N x K operand)
//   C/D (16x16 f32): lane L gives N=L%16, VGPR v gives M=8*(L/16)+v
// ---------------------------------------------------------------------------
union ABf { v16bf v; v8bf h[2]; };

__device__ inline v8f wmma_tile_k(const __bf16* __restrict__ arow,
                                  const __bf16* __restrict__ brow,
                                  int K, int hi, v8f acc)
{
    for (int k0 = 0; k0 < K; k0 += 32) {
        ABf A, Bm;
        A.h[0]  = *(const v8bf*)(arow + k0 + hi * 8);
        A.h[1]  = *(const v8bf*)(arow + k0 + 16 + hi * 8);
        Bm.h[0] = *(const v8bf*)(brow + k0 + hi * 8);
        Bm.h[1] = *(const v8bf*)(brow + k0 + 16 + hi * 8);
        acc = __builtin_amdgcn_wmma_f32_16x16x32_bf16(
                  false, A.v, false, Bm.v, (short)0, acc, false, false);
    }
    return acc;
}

// Generic: Y[M,Nc] = act(X[M,K] * W + bias), fp32 out (value/policy heads)
__global__ __launch_bounds__(32)
void wmma_gemm_kernel(const __bf16* __restrict__ X, const __bf16* __restrict__ Wt,
                      const float* __restrict__ bias, float* __restrict__ Y,
                      int K, int Nc, int relu)
{
    const int lane = threadIdx.x, r = lane & 15, hi = lane >> 4;
    const int m0 = blockIdx.x << 4, n0 = blockIdx.y << 4;
    v8f acc = {};
    acc = wmma_tile_k(X + (size_t)(m0 + r) * K, Wt + (size_t)(n0 + r) * K, K, hi, acc);
    const float bn = bias ? bias[n0 + r] : 0.0f;
#pragma unroll
    for (int v = 0; v < 8; ++v) {
        float y = acc[v] + bn;
        if (relu) y = fmaxf(y, 0.0f);
        Y[(size_t)(m0 + hi * 8 + v) * Nc + (n0 + r)] = y;
    }
}

// GCN stage 1: hwT[b][n][m_local] = (X @ W)[m,n] stored bf16, per-graph
// transposed (HD x 64) -- exactly the N x K operand layout the aggregation
// GEMM's B-fragment loader consumes.
__global__ __launch_bounds__(32)
void xw_gemm_kernel(const __bf16* __restrict__ X, const __bf16* __restrict__ Wt,
                    __bf16* __restrict__ hwT, int K)
{
    const int lane = threadIdx.x, r = lane & 15, hi = lane >> 4;
    const int m0 = blockIdx.x << 4, n0 = blockIdx.y << 4;
    v8f acc = {};
    acc = wmma_tile_k(X + (size_t)(m0 + r) * K, Wt + (size_t)(n0 + r) * K, K, hi, acc);
#pragma unroll
    for (int v = 0; v < 8; ++v) {
        int m = m0 + hi * 8 + v;                 // global node
        int b = m >> 6, ml = m & (NNODE - 1);    // graph, local node
        hwT[(size_t)b * HD * NNODE + (size_t)(n0 + r) * NNODE + ml] = (__bf16)acc[v];
    }
}

// GCN stage 2: h = relu(Aadj[b] @ hwT[b]^T + bias); dual fp32/bf16 store.
// Aadj[b] is 64x64 bf16 (row dst, col src) with dinv^2 self-loop on diagonal.
__global__ __launch_bounds__(32)
void agg_gemm_kernel(const __bf16* __restrict__ Aadj, const __bf16* __restrict__ hwT,
                     const float* __restrict__ bias,
                     float* __restrict__ h, __bf16* __restrict__ hbf)
{
    const int lane = threadIdx.x, r = lane & 15, hi = lane >> 4;
    const int m0 = blockIdx.x << 4, n0 = blockIdx.y << 4;
    const int b = m0 >> 6, ml = m0 & (NNODE - 1);
    const __bf16* arow = Aadj + (size_t)b * NNODE * NNODE + (size_t)(ml + r) * NNODE;
    const __bf16* brow = hwT  + (size_t)b * HD * NNODE    + (size_t)(n0 + r) * NNODE;
    v8f acc = {};
    acc = wmma_tile_k(arow, brow, NNODE, hi, acc);   // K = 64 nodes
    const float bn = bias[n0 + r];
#pragma unroll
    for (int v = 0; v < 8; ++v) {
        float y = fmaxf(acc[v] + bn, 0.0f);
        size_t idx = (size_t)(m0 + hi * 8 + v) * HD + (n0 + r);
        h[idx]   = y;
        hbf[idx] = (__bf16)y;
    }
}

// ---------------------------------------------------------------------------
// Helpers
// ---------------------------------------------------------------------------
__global__ void cast_bf16_kernel(const float* __restrict__ in, __bf16* __restrict__ out, int n)
{
    int i = blockIdx.x * blockDim.x + threadIdx.x;
    if (i < n) out[i] = (__bf16)in[i];
}

// W: K x Nc row-major fp32 -> Wt: Nc x K row-major bf16
__global__ void cast_transpose_kernel(const float* __restrict__ W, __bf16* __restrict__ Wt,
                                      int K, int Nc)
{
    int i = blockIdx.x * blockDim.x + threadIdx.x;
    if (i < K * Nc) {
        int k = i / Nc, n = i - k * Nc;
        Wt[(size_t)n * K + k] = (__bf16)W[i];
    }
}

__global__ void zero_kernel(float* __restrict__ p, int n)
{
    int i = blockIdx.x * blockDim.x + threadIdx.x;
    if (i < n) p[i] = 0.0f;
}

__global__ void degree_kernel(const int* __restrict__ dst, float* __restrict__ deg)
{
    int e = blockIdx.x * blockDim.x + threadIdx.x;
    if (e < EDGES) atomicAdd(&deg[dst[e]], 1.0f);
}

__global__ void dinv_kernel(const float* __restrict__ deg, float* __restrict__ dinv)
{
    int i = blockIdx.x * blockDim.x + threadIdx.x;
    if (i < NNTOT) dinv[i] = rsqrtf(deg[i] + 1.0f);
}

// Dense normalized adjacency: Aadjf[b][dst][src] += dinv_s*dinv_d per edge
__global__ void edge_adj_kernel(const int* __restrict__ src, const int* __restrict__ dst,
                                const float* __restrict__ dinv, float* __restrict__ Aadjf)
{
    int e = blockIdx.x * blockDim.x + threadIdx.x;
    if (e < EDGES) {
        int s = src[e], d = dst[e];
        int b = d >> 6;
        atomicAdd(&Aadjf[(size_t)b * NNODE * NNODE + (size_t)(d & 63) * NNODE + (s & 63)],
                  dinv[s] * dinv[d]);
    }
}

// Self-loop term on the diagonal: Aadjf[b][i][i] += dinv[i]^2
__global__ void diag_adj_kernel(const float* __restrict__ dinv, float* __restrict__ Aadjf)
{
    int i = blockIdx.x * blockDim.x + threadIdx.x;
    if (i < NNTOT) {
        int b = i >> 6, l = i & 63;
        float di = dinv[i];
        Aadjf[(size_t)b * NNODE * NNODE + (size_t)l * NNODE + l] += di * di;
    }
}

// mean pool over the 64 nodes of each graph
__global__ __launch_bounds__(HD)
void pool_kernel(const float* __restrict__ h, float* __restrict__ pooled,
                 __bf16* __restrict__ pooledbf)
{
    int b = blockIdx.x, t = threadIdx.x;
    float s = 0.0f;
    for (int n = 0; n < NNODE; ++n)
        s += h[((size_t)(b * NNODE + n)) * HD + t];
    s *= (1.0f / NNODE);
    pooled[b * HD + t]   = s;
    pooledbf[b * HD + t] = (__bf16)s;
}

// v[b] = tanh( relu_out[b,:] . Wv2 + bv2 )
__global__ __launch_bounds__(64)
void value_kernel(const float* __restrict__ t1, const float* __restrict__ wv2,
                  const float* __restrict__ bv2, float* __restrict__ vout)
{
    int b = threadIdx.x;
    float s = 0.0f;
    for (int h = 0; h < HD; ++h) s += t1[b * HD + h] * wv2[h];
    vout[b] = tanhf(s + bv2[0]);
}

// pi[b,i,j] = adj ? sum_h relu(a[b,i,h] + c[b,j,h]) * wpi2[h] + bpi2 : 0
// (b_pi1 folded into a's GEMM bias). Block per (b,i); 8 waves over j;
// wave32 shfl_xor reduction over h.
__global__ __launch_bounds__(256)
void policy_kernel(const float* __restrict__ a, const float* __restrict__ c,
                   const float* __restrict__ wpi2, const float* __restrict__ bpi2,
                   const unsigned char* __restrict__ adj, float* __restrict__ pi)
{
    __shared__ float as[HD];
    __shared__ float wsh[HD];
    int bi = blockIdx.x;          // b*64 + i
    int b  = bi >> 6;
    int t  = threadIdx.x;
    as[t]  = a[(size_t)bi * HD + t];
    wsh[t] = wpi2[t];
    __syncthreads();

    int wave = t >> 5, lane = t & 31;
    float bp = bpi2[0];
    for (int j = wave; j < NNODE; j += 8) {
        const float* crow = c + ((size_t)(b * NNODE + j)) * HD;
        float sum = 0.0f;
#pragma unroll
        for (int hh = 0; hh < HD / 32; ++hh) {
            int h = lane + hh * 32;
            sum += fmaxf(as[h] + crow[h], 0.0f) * wsh[h];
        }
        for (int off = 16; off > 0; off >>= 1)
            sum += __shfl_xor(sum, off, 32);
        if (lane == 0) {
            size_t idx = (size_t)bi * NNODE + j;
            pi[idx] = adj[idx] ? (sum + bp) : 0.0f;
        }
    }
}

// ---------------------------------------------------------------------------
// Host launch
// ---------------------------------------------------------------------------
extern "C" void kernel_launch(void* const* d_in, const int* in_sizes, int n_in,
                              void* d_out, int out_size, void* d_ws, size_t ws_size,
                              hipStream_t stream)
{
    const float* x    = (const float*)d_in[0];
    const float* W0   = (const float*)d_in[1];
    const float* b0   = (const float*)d_in[2];
    const float* W1   = (const float*)d_in[3];
    const float* b1   = (const float*)d_in[4];
    const float* W2   = (const float*)d_in[5];
    const float* b2   = (const float*)d_in[6];
    const float* Wpi1 = (const float*)d_in[7];   // (512,256): Wa rows 0..255, Wb rows 256..511
    const float* bpi1 = (const float*)d_in[8];
    const float* Wpi2 = (const float*)d_in[9];
    const float* bpi2 = (const float*)d_in[10];
    const float* Wv1  = (const float*)d_in[11];
    const float* bv1  = (const float*)d_in[12];
    const float* Wv2  = (const float*)d_in[13];
    const float* bv2  = (const float*)d_in[14];
    const int*   eidx = (const int*)d_in[15];
    const unsigned char* adj = (const unsigned char*)d_in[17];  // bool move_adj

    const int* srcI = eidx;
    const int* dstI = eidx + EDGES;

    float* pi   = (float*)d_out;                               // 64*64*64
    float* vout = (float*)d_out + (size_t)BB * NNODE * NNODE;  // 64

    // --- carve workspace ---
    uintptr_t base = (uintptr_t)d_ws;
    size_t off = 0;
    auto carve = [&](size_t bytes) -> void* {
        off = (off + 255) & ~(size_t)255;
        void* p = (void*)(base + off);
        off += bytes;
        return p;
    };
    __bf16* x_bf    = (__bf16*)carve((size_t)NNTOT * FIN * 2);
    __bf16* W0t     = (__bf16*)carve((size_t)HD * FIN * 2);
    __bf16* W1t     = (__bf16*)carve((size_t)HD * HD * 2);
    __bf16* W2t     = (__bf16*)carve((size_t)HD * HD * 2);
    __bf16* Wat     = (__bf16*)carve((size_t)HD * HD * 2);
    __bf16* Wbt     = (__bf16*)carve((size_t)HD * HD * 2);
    __bf16* Wv1t    = (__bf16*)carve((size_t)HD * HD * 2);
    __bf16* hwT     = (__bf16*)carve((size_t)BB * HD * NNODE * 2);   // per-graph (HD x 64)
    float*  h       = (float*)carve((size_t)NNTOT * HD * 4);
    __bf16* hbf     = (__bf16*)carve((size_t)NNTOT * HD * 2);
    float*  deg     = (float*)carve((size_t)NNTOT * 4);
    float*  dinv    = (float*)carve((size_t)NNTOT * 4);
    float*  Aadjf   = (float*)carve((size_t)BB * NNODE * NNODE * 4);
    __bf16* Aadj    = (__bf16*)carve((size_t)BB * NNODE * NNODE * 2);
    float*  abuf    = (float*)carve((size_t)NNTOT * HD * 4);
    float*  cbuf    = (float*)carve((size_t)NNTOT * HD * 4);
    float*  pooled  = (float*)carve((size_t)BB * HD * 4);
    __bf16* pooledbf = (__bf16*)carve((size_t)BB * HD * 2);
    float*  vtmp    = (float*)carve((size_t)BB * HD * 4);

    const int T = 256;
    const int ADJN = BB * NNODE * NNODE;   // 262144

    // weights -> transposed bf16
    cast_transpose_kernel<<<(FIN * HD + T - 1) / T, T, 0, stream>>>(W0, W0t, FIN, HD);
    cast_transpose_kernel<<<(HD * HD + T - 1) / T, T, 0, stream>>>(W1, W1t, HD, HD);
    cast_transpose_kernel<<<(HD * HD + T - 1) / T, T, 0, stream>>>(W2, W2t, HD, HD);
    cast_transpose_kernel<<<(HD * HD + T - 1) / T, T, 0, stream>>>(Wpi1, Wat, HD, HD);
    cast_transpose_kernel<<<(HD * HD + T - 1) / T, T, 0, stream>>>(Wpi1 + (size_t)HD * HD, Wbt, HD, HD);
    cast_transpose_kernel<<<(HD * HD + T - 1) / T, T, 0, stream>>>(Wv1, Wv1t, HD, HD);
    cast_bf16_kernel<<<(NNTOT * FIN + T - 1) / T, T, 0, stream>>>(x, x_bf, NNTOT * FIN);

    // normalization + dense per-graph adjacency (built once, reused 3 layers)
    zero_kernel<<<(NNTOT + T - 1) / T, T, 0, stream>>>(deg, NNTOT);
    degree_kernel<<<(EDGES + T - 1) / T, T, 0, stream>>>(dstI, deg);
    dinv_kernel<<<(NNTOT + T - 1) / T, T, 0, stream>>>(deg, dinv);
    zero_kernel<<<(ADJN + T - 1) / T, T, 0, stream>>>(Aadjf, ADJN);
    edge_adj_kernel<<<(EDGES + T - 1) / T, T, 0, stream>>>(srcI, dstI, dinv, Aadjf);
    diag_adj_kernel<<<(NNTOT + T - 1) / T, T, 0, stream>>>(dinv, Aadjf);
    cast_bf16_kernel<<<(ADJN + T - 1) / T, T, 0, stream>>>(Aadjf, Aadj, ADJN);

    dim3 gBig(NNTOT / 16, HD / 16);   // 256 x 16 wavetiles

    // --- 3 GCN layers: h' = relu(Aadj_b @ (h @ W)_b + bias), all-WMMA ---
    xw_gemm_kernel<<<gBig, 32, 0, stream>>>(x_bf, W0t, hwT, FIN);
    agg_gemm_kernel<<<gBig, 32, 0, stream>>>(Aadj, hwT, b0, h, hbf);

    xw_gemm_kernel<<<gBig, 32, 0, stream>>>(hbf, W1t, hwT, HD);
    agg_gemm_kernel<<<gBig, 32, 0, stream>>>(Aadj, hwT, b1, h, hbf);

    xw_gemm_kernel<<<gBig, 32, 0, stream>>>(hbf, W2t, hwT, HD);
    agg_gemm_kernel<<<gBig, 32, 0, stream>>>(Aadj, hwT, b2, h, hbf);

    // --- value head ---
    pool_kernel<<<BB, HD, 0, stream>>>(h, pooled, pooledbf);
    dim3 gVal(BB / 16, HD / 16);
    wmma_gemm_kernel<<<gVal, 32, 0, stream>>>(pooledbf, Wv1t, bv1, vtmp, HD, HD, 1);
    value_kernel<<<1, 64, 0, stream>>>(vtmp, Wv2, bv2, vout);

    // --- policy head: a = h@Wa + b_pi1, c = h@Wb ---
    wmma_gemm_kernel<<<gBig, 32, 0, stream>>>(hbf, Wat, bpi1, abuf, HD, HD, 0);
    wmma_gemm_kernel<<<gBig, 32, 0, stream>>>(hbf, Wbt, nullptr, cbuf, HD, HD, 0);
    policy_kernel<<<NNTOT, 256, 0, stream>>>(abuf, cbuf, Wpi2, bpi2, adj, pi);

    (void)in_sizes; (void)n_in; (void)out_size; (void)ws_size;
}
